// AGCLSTM_10101763080581
// MI455X (gfx1250) — matrix operand order repaired and checked by reading, not compiled, not run on hardware
//
#include <hip/hip_runtime.h>
#include <hip/hip_bf16.h>

// AGC-LSTM for MI455X (gfx1250, wave32, WMMA).
// Heavy path: per-step (32 x 4096) @ (4096 x 15024) GEMM via v_wmma_f32_16x16x32_bf16.
// Weights converted once per launch to bf16, pre-packed in WMMA B-fragment order so the
// packed image (123 MB) is L2-resident (192 MB) across the 12 sequential LSTM steps.
// GEMM: one wave per 16-col N-tile computes BOTH 16-row M-tiles (M=32), so each B
// fragment is loaded once and feeds two independent WMMA accumulation chains; fragment
// loads are double-buffered so WMMA overlaps the in-flight loads of the next k-tile.

namespace {

constexpr int BSZ = 32, NNOD = 24, TSTP = 12, FIN = 3, KC = 3, CO = 12;
constexpr int DIN  = NNOD*CO + NNOD + 1;   // 313
constexpr int HH   = DIN*TSTP;             // 3756
constexpr int H4   = 4*HH;                 // 15024
constexpr int KPAD = 4096;                 // 313 + 3756 = 4069 -> pad
constexpr int KT   = KPAD/32;              // 128 k-tiles
constexpr int NTI  = H4/16;                // 939 n-tiles
constexpr int MTI  = 2;                    // 32 rows / 16
constexpr float LNE = 1e-5f;

typedef __bf16 v16bf __attribute__((ext_vector_type(16)));
typedef float  v8f   __attribute__((ext_vector_type(8)));

union FragU { uint4 q[2]; v16bf v; };

__device__ __forceinline__ float sigf(float x){ return 1.0f/(1.0f + expf(-x)); }

__device__ __forceinline__ unsigned short f2bf(float f){
  unsigned int u = __float_as_uint(f);
  unsigned int r = u + 0x7FFFu + ((u >> 16) & 1u);   // round-to-nearest-even
  return (unsigned short)(r >> 16);
}

// ---------------------------------------------------------------------------
// Weight pack: Wcat[j,k] = k<DIN ? W_ih[j,k] : (k<DIN+HH ? W_hh[j,k-DIN] : 0)
// B-fragment (32x16 bf16, this code's contiguous k order, same as A): lane -> n=lane&15, half=lane>>4,
// dword v in 0..7 packs K = kt*32 + half*16 + v*2 (+1). 1 KB per fragment,
// fragment index = nt*KT + kt, lane-contiguous 32B blocks.
__global__ void agc_packW(const float* __restrict__ Wih, const float* __restrict__ Whh,
                          unsigned int* __restrict__ Wpk)
{
  size_t tid = (size_t)blockIdx.x*256 + threadIdx.x;
  const size_t TOT = (size_t)NTI*KT*32*8;
  if (tid >= TOT) return;
  int v    = (int)(tid & 7);
  int lane = (int)((tid >> 3) & 31);
  size_t f = tid >> 8;                  // nt*KT + kt
  int kt = (int)(f & (size_t)(KT-1));
  int nt = (int)(f >> 7);
  int j  = nt*16 + (lane & 15);
  int kb = kt*32 + (lane >> 4)*16 + v*2;
  float x0 = 0.f, x1 = 0.f;
  {
    int k = kb;
    x0 = (k < DIN) ? Wih[(size_t)j*DIN + k]
       : (k < DIN+HH) ? Whh[(size_t)j*HH + (k - DIN)] : 0.f;
    k = kb + 1;
    x1 = (k < DIN) ? Wih[(size_t)j*DIN + k]
       : (k < DIN+HH) ? Whh[(size_t)j*HH + (k - DIN)] : 0.f;
  }
  Wpk[tid] = (unsigned int)f2bf(x0) | ((unsigned int)f2bf(x1) << 16);
}

// A-fragment (16x32 bf16): lane -> m=lane&15, half=lane>>4,
// dword v packs K = kt*32 + half*16 + v*2 (+1).
__global__ void agc_packA(const float* __restrict__ Af, unsigned int* __restrict__ Apk)
{
  int tid = blockIdx.x*256 + threadIdx.x;   // MTI*KT*32*8 = 65536
  int v    = tid & 7;
  int lane = (tid >> 3) & 31;
  int f    = tid >> 8;                      // mt*KT + kt
  int kt = f & (KT-1);
  int mt = f >> 7;
  int m  = mt*16 + (lane & 15);
  int kb = kt*32 + (lane >> 4)*16 + v*2;
  float x0 = Af[(size_t)m*KPAD + kb];
  float x1 = Af[(size_t)m*KPAD + kb + 1];
  Apk[tid] = (unsigned int)f2bf(x0) | ((unsigned int)f2bf(x1) << 16);
}

// gates = Acat @ Wcat^T + b_ih + b_hh.
// One wave per N-tile: 2 accumulators (m=0..15, m=16..31), B fragment shared,
// double-buffered fragment loads, 2x manual unroll so all buffers stay in registers.
__global__ void __launch_bounds__(256)
agc_gates(const unsigned short* __restrict__ Apk, const unsigned short* __restrict__ Wpk,
          const float* __restrict__ bih, const float* __restrict__ bhh,
          float* __restrict__ gates)
{
  int lane = threadIdx.x & 31;
  int wave = threadIdx.x >> 5;
  int nt = blockIdx.x*8 + wave;
  if (nt >= NTI) return;                 // wave-uniform: EXEC stays all-1s for WMMA
  const unsigned short* a0 = Apk + lane*16;                       // mt=0 fragments
  const unsigned short* a1 = Apk + (size_t)KT*512 + lane*16;      // mt=1 fragments
  const unsigned short* bB = Wpk + (size_t)nt*KT*512 + lane*16;   // B fragments

  v8f acc0 = {}, acc1 = {};
  FragU fbA, f0A, f1A, fbB, f0B, f1B;

  auto ld = [&](FragU& d, const unsigned short* base, int kt) {
    const uint4* q = (const uint4*)(base + (size_t)kt*512);
    d.q[0] = q[0]; d.q[1] = q[1];
  };

  // Preload k-tile 0 into buffer A.
  ld(fbA, bB, 0); ld(f0A, a0, 0); ld(f1A, a1, 0);

  for (int kt = 0; kt < KT; kt += 2) {
    __builtin_prefetch((const void*)(bB + (size_t)(kt + 8)*512), 0, 3);
    // Issue loads for kt+1 (always valid: KT even), then WMMA on buffer A.
    ld(fbB, bB, kt + 1); ld(f0B, a0, kt + 1); ld(f1B, a1, kt + 1);
    acc0 = __builtin_amdgcn_wmma_f32_16x16x32_bf16(
        false, f0A.v, false, fbA.v, (short)0, acc0, false, false);
    acc1 = __builtin_amdgcn_wmma_f32_16x16x32_bf16(
        false, f1A.v, false, fbA.v, (short)0, acc1, false, false);
    // Issue loads for kt+2, then WMMA on buffer B.
    if (kt + 2 < KT) { ld(fbA, bB, kt + 2); ld(f0A, a0, kt + 2); ld(f1A, a1, kt + 2); }
    acc0 = __builtin_amdgcn_wmma_f32_16x16x32_bf16(
        false, f0B.v, false, fbB.v, (short)0, acc0, false, false);
    acc1 = __builtin_amdgcn_wmma_f32_16x16x32_bf16(
        false, f1B.v, false, fbB.v, (short)0, acc1, false, false);
  }

  int j  = nt*16 + (lane & 15);
  int mb = (lane >> 4)*8;                // C/D layout: lanes 0-15 -> M=r, 16-31 -> M=8+r
  float bias = bih[j] + bhh[j];
  #pragma unroll
  for (int r = 0; r < 8; ++r) {
    gates[(size_t)(mb + r)*H4 + j]      = acc0[r] + bias;
    gates[(size_t)(16 + mb + r)*H4 + j] = acc1[r] + bias;
  }
}

// ---------------------------------------------------------------------------
// Init: Acat rows [xi(313) | h(3756) | 0 pad(27)], c = c0.
__global__ void agc_init(const float* __restrict__ h0, const float* __restrict__ c0,
                         float* __restrict__ Af, float* __restrict__ cbuf)
{
  int tid = blockIdx.x*256 + threadIdx.x;
  if (tid < BSZ*KPAD) {
    int b = tid / KPAD, k = tid % KPAD;
    float v = 0.f;
    if (k >= DIN && k < DIN + HH) v = h0[(size_t)b*HH + (k - DIN)];
    Af[tid] = v;
  }
  if (tid < BSZ*HH) cbuf[tid] = c0[tid];
}

// Per-step: alpha = softmax(sig(x_t @ sa2_w^T + sa2_b)), xi = x_t*alpha + x_t.
__global__ void __launch_bounds__(512)
agc_alpha(const float* __restrict__ x_in, const float* __restrict__ sa2w,
          const float* __restrict__ sa2b, float* __restrict__ Af, int t)
{
  __shared__ float xs[DIN];
  __shared__ float red[512];
  int b = blockIdx.x, d = threadIdx.x;
  if (d < DIN) xs[d] = x_in[((size_t)b*DIN + d)*TSTP + t];
  __syncthreads();
  float s = -1e30f;
  if (d < DIN) {
    float acc = sa2b[d];
    const float* wr = sa2w + (size_t)d*DIN;
    for (int k = 0; k < DIN; ++k) acc += xs[k]*wr[k];
    s = sigf(acc);
  }
  red[d] = s; __syncthreads();
  for (int st = 256; st > 0; st >>= 1) { if (d < st) red[d] = fmaxf(red[d], red[d+st]); __syncthreads(); }
  float mx = red[0]; __syncthreads();
  float e = (d < DIN) ? expf(s - mx) : 0.f;
  red[d] = e; __syncthreads();
  for (int st = 256; st > 0; st >>= 1) { if (d < st) red[d] += red[d+st]; __syncthreads(); }
  float sum = red[0];
  if (d < DIN) Af[(size_t)b*KPAD + d] = xs[d]*(e/sum) + xs[d];
}

__global__ void agc_cell(const float* __restrict__ gates, float* __restrict__ cbuf,
                         float* __restrict__ hs, float* __restrict__ Af, int t)
{
  int tid = blockIdx.x*256 + threadIdx.x;
  if (tid >= BSZ*HH) return;
  int b = tid / HH, h = tid % HH;
  const float* g = gates + (size_t)b*H4;
  float ig = sigf(g[h]);
  float fg = sigf(g[HH + h]);
  float gg = tanhf(g[2*HH + h]);
  float og = sigf(g[3*HH + h]);
  float c2 = fg*cbuf[tid] + ig*gg;
  float h2 = og*tanhf(c2);
  cbuf[tid] = c2;
  hs[((size_t)t*BSZ + b)*HH + h] = h2;
  Af[(size_t)b*KPAD + DIN + h] = h2;
}

// ---------------------------------------------------------------------------
// Spatial attention. X is (B,N,T,F) row-major.
__global__ void agc_sa1(const float* __restrict__ X, const float* __restrict__ W1,
                        const float* __restrict__ W2, const float* __restrict__ W3,
                        float* __restrict__ lhs, float* __restrict__ rhs)
{
  int tid = blockIdx.x*256 + threadIdx.x;             // B*N*T
  if (tid >= BSZ*NNOD*TSTP) return;
  int t = tid % TSTP, n = (tid/TSTP) % NNOD, b = tid/(TSTP*NNOD);
  const float* xb = X + ((size_t)b*NNOD + n)*TSTP*FIN;
  float acc = 0.f;
  for (int f = 0; f < FIN; ++f) {
    float inner = 0.f;
    for (int t2 = 0; t2 < TSTP; ++t2) inner += xb[t2*FIN + f]*W1[t2];
    acc += inner*W2[f*TSTP + t];
  }
  lhs[((size_t)b*NNOD + n)*TSTP + t] = acc;
  float r = 0.f;
  for (int f = 0; f < FIN; ++f) r += W3[f]*xb[t*FIN + f];
  rhs[((size_t)b*TSTP + t)*NNOD + n] = r;
}

__global__ void agc_sa2(const float* __restrict__ lhs, const float* __restrict__ rhs,
                        const float* __restrict__ bs, float* __restrict__ Q)
{
  int tid = blockIdx.x*256 + threadIdx.x;             // B*N*N : (b,m,k)
  if (tid >= BSZ*NNOD*NNOD) return;
  int k = tid % NNOD, m = (tid/NNOD) % NNOD, b = tid/(NNOD*NNOD);
  float acc = bs[m*NNOD + k];
  for (int t = 0; t < TSTP; ++t)
    acc += lhs[((size_t)b*NNOD + m)*TSTP + t]*rhs[((size_t)b*TSTP + t)*NNOD + k];
  Q[tid] = sigf(acc);
}

__global__ void agc_sa3(const float* __restrict__ Vs, const float* __restrict__ Q,
                        float* __restrict__ S)
{
  int tid = blockIdx.x*256 + threadIdx.x;             // (b,n,k)
  if (tid >= BSZ*NNOD*NNOD) return;
  int k = tid % NNOD, n = (tid/NNOD) % NNOD, b = tid/(NNOD*NNOD);
  float acc = 0.f;
  for (int m = 0; m < NNOD; ++m)
    acc += Vs[n*NNOD + m]*Q[((size_t)b*NNOD + m)*NNOD + k];
  S[tid] = acc;
}

__global__ void agc_sa4(const float* __restrict__ S, float* __restrict__ q2)
{
  int tid = blockIdx.x*256 + threadIdx.x;             // (b,k)
  if (tid >= BSZ*NNOD) return;
  int k = tid % NNOD, b = tid/NNOD;
  float mx = -1e30f;
  for (int n = 0; n < NNOD; ++n) mx = fmaxf(mx, S[((size_t)b*NNOD + n)*NNOD + k]);
  float sum = 0.f;
  for (int n = 0; n < NNOD; ++n) sum += expf(S[((size_t)b*NNOD + n)*NNOD + k] - mx);
  for (int n = 0; n < NNOD; ++n) {
    float sm = expf(S[((size_t)b*NNOD + n)*NNOD + k] - mx)/sum;
    q2[((size_t)b*NNOD + n)*NNOD + k] = sigf(sm);
  }
}

__global__ void agc_sa5(const float* __restrict__ q2, float* __restrict__ sa,
                        float* __restrict__ sa_out)
{
  int tid = blockIdx.x*256 + threadIdx.x;             // (n,k)
  if (tid >= NNOD*NNOD) return;
  int k = tid % NNOD, n = tid/NNOD;
  float mx = -1e30f;
  for (int b = 0; b < BSZ; ++b) mx = fmaxf(mx, q2[((size_t)b*NNOD + n)*NNOD + k]);
  float sum = 0.f;
  for (int b = 0; b < BSZ; ++b) sum += expf(q2[((size_t)b*NNOD + n)*NNOD + k] - mx);
  for (int b = 0; b < BSZ; ++b) {
    float v = expf(q2[((size_t)b*NNOD + n)*NNOD + k] - mx)/sum;
    sa[((size_t)b*NNOD + n)*NNOD + k] = v;
    sa_out[((size_t)b*NNOD + n)*NNOD + k] = v;
  }
}

// Temporal attention.
__global__ void agc_ta1(const float* __restrict__ X, const float* __restrict__ U1,
                        const float* __restrict__ U2, const float* __restrict__ U3,
                        float* __restrict__ lhs_t, float* __restrict__ rhs_t)
{
  int tid = blockIdx.x*256 + threadIdx.x;             // (b,n,t)
  if (tid >= BSZ*NNOD*TSTP) return;
  int t = tid % TSTP, n = (tid/TSTP) % NNOD, b = tid/(TSTP*NNOD);
  float acc = 0.f;
  for (int f = 0; f < FIN; ++f) {
    float inner = 0.f;
    for (int m = 0; m < NNOD; ++m)
      inner += X[((size_t)b*NNOD + m)*TSTP*FIN + t*FIN + f]*U1[m];
    acc += inner*U2[f*NNOD + n];
  }
  lhs_t[((size_t)b*TSTP + t)*NNOD + n] = acc;
  float r = 0.f;
  for (int f = 0; f < FIN; ++f)
    r += U3[f]*X[((size_t)b*NNOD + n)*TSTP*FIN + t*FIN + f];
  rhs_t[((size_t)b*NNOD + n)*TSTP + t] = r;
}

__global__ void agc_ta2(const float* __restrict__ lhs_t, const float* __restrict__ rhs_t,
                        const float* __restrict__ be, float* __restrict__ QE)
{
  int tid = blockIdx.x*256 + threadIdx.x;             // (b,t,u)
  if (tid >= BSZ*TSTP*TSTP) return;
  int u = tid % TSTP, t = (tid/TSTP) % TSTP, b = tid/(TSTP*TSTP);
  float acc = be[t*TSTP + u];
  for (int n = 0; n < NNOD; ++n)
    acc += lhs_t[((size_t)b*TSTP + t)*NNOD + n]*rhs_t[((size_t)b*NNOD + n)*TSTP + u];
  QE[tid] = sigf(acc);
}

__global__ void agc_ta3(const float* __restrict__ Ve, const float* __restrict__ QE,
                        float* __restrict__ E)
{
  int tid = blockIdx.x*256 + threadIdx.x;             // (b,t,v)
  if (tid >= BSZ*TSTP*TSTP) return;
  int v = tid % TSTP, t = (tid/TSTP) % TSTP, b = tid/(TSTP*TSTP);
  float acc = 0.f;
  for (int u = 0; u < TSTP; ++u)
    acc += Ve[t*TSTP + u]*QE[((size_t)b*TSTP + u)*TSTP + v];
  E[tid] = acc;
}

__global__ void agc_ta4(const float* __restrict__ E, float* __restrict__ q2)
{
  int tid = blockIdx.x*256 + threadIdx.x;             // (b,v)
  if (tid >= BSZ*TSTP) return;
  int v = tid % TSTP, b = tid/TSTP;
  float mx = -1e30f;
  for (int t = 0; t < TSTP; ++t) mx = fmaxf(mx, E[((size_t)b*TSTP + t)*TSTP + v]);
  float sum = 0.f;
  for (int t = 0; t < TSTP; ++t) sum += expf(E[((size_t)b*TSTP + t)*TSTP + v] - mx);
  for (int t = 0; t < TSTP; ++t)
    q2[((size_t)b*TSTP + t)*TSTP + v] = sigf(expf(E[((size_t)b*TSTP + t)*TSTP + v] - mx)/sum);
}

__global__ void agc_ta5(const float* __restrict__ q2, float* __restrict__ ta)
{
  int tid = blockIdx.x*256 + threadIdx.x;             // (t,v)
  if (tid >= TSTP*TSTP) return;
  int v = tid % TSTP, t = tid/TSTP;
  float mx = -1e30f;
  for (int b = 0; b < BSZ; ++b) mx = fmaxf(mx, q2[((size_t)b*TSTP + t)*TSTP + v]);
  float sum = 0.f;
  for (int b = 0; b < BSZ; ++b) sum += expf(q2[((size_t)b*TSTP + t)*TSTP + v] - mx);
  for (int b = 0; b < BSZ; ++b)
    ta[((size_t)b*TSTP + t)*TSTP + v] = expf(q2[((size_t)b*TSTP + t)*TSTP + v] - mx)/sum;
}

// Chebyshev graph conv. xg is raw reshape of X: xg[b,m,f,t] = Xflat[(b*N+m)*36 + f*12 + t].
__global__ void agc_rg(const float* __restrict__ X, const float* __restrict__ cheb,
                       const float* __restrict__ sa, float* __restrict__ rg)
{
  int tid = blockIdx.x*256 + threadIdx.x;             // (b,k,n,f,t)
  if (tid >= BSZ*KC*NNOD*FIN*TSTP) return;
  int t = tid % TSTP, f = (tid/TSTP) % FIN, n = (tid/(TSTP*FIN)) % NNOD;
  int k = (tid/(TSTP*FIN*NNOD)) % KC, b = tid/(TSTP*FIN*NNOD*KC);
  float acc = 0.f;
  for (int m = 0; m < NNOD; ++m)
    acc += cheb[(k*NNOD + m)*NNOD + n]*sa[((size_t)b*NNOD + m)*NNOD + n]
         * X[((size_t)b*NNOD + m)*36 + f*12 + t];
  rg[tid] = acc;
}

__global__ void agc_gc(const float* __restrict__ rg, const float* __restrict__ theta,
                       const float* __restrict__ cb, float* __restrict__ xc)
{
  int tid = blockIdx.x*256 + threadIdx.x;             // (b,n,o,t)
  if (tid >= BSZ*NNOD*CO*TSTP) return;
  int t = tid % TSTP, o = (tid/TSTP) % CO, n = (tid/(TSTP*CO)) % NNOD, b = tid/(TSTP*CO*NNOD);
  float acc = cb[o];
  for (int k = 0; k < KC; ++k)
    for (int f = 0; f < FIN; ++f)
      acc += theta[(k*FIN + f)*CO + o]*rg[((((size_t)b*KC + k)*NNOD + n)*FIN + f)*TSTP + t];
  xc[tid] = fmaxf(acc, 0.f);
}

// LayerNorm over T; note ln_g/ln_b broadcast along the LAST axis (t), since COUT==T.
__global__ void agc_ln(const float* __restrict__ xc, const float* __restrict__ lng,
                       const float* __restrict__ lnb, float* __restrict__ xgc)
{
  int tid = blockIdx.x*256 + threadIdx.x;             // (b,n,o) rows
  if (tid >= BSZ*NNOD*CO) return;
  const float* row = xc + (size_t)tid*TSTP;
  float mu = 0.f;
  for (int t = 0; t < TSTP; ++t) mu += row[t];
  mu *= (1.f/TSTP);
  float var = 0.f;
  for (int t = 0; t < TSTP; ++t) { float d = row[t] - mu; var += d*d; }
  var *= (1.f/TSTP);
  float inv = rsqrtf(var + LNE);
  for (int t = 0; t < TSTP; ++t)
    xgc[(size_t)tid*TSTP + t] = (row[t] - mu)*inv*lng[t] + lnb[t];
}

__global__ void agc_xgct(const float* __restrict__ xgc, const float* __restrict__ ta,
                         float* __restrict__ xres)
{
  int tid = blockIdx.x*256 + threadIdx.x;             // (b, i<288, k)
  if (tid >= BSZ*(NNOD*CO)*TSTP) return;
  int k = tid % TSTP, i = (tid/TSTP) % (NNOD*CO), b = tid/(TSTP*NNOD*CO);
  float acc = 0.f;
  for (int j = 0; j < TSTP; ++j)
    acc += xgc[((size_t)b*(NNOD*CO) + i)*TSTP + j]*ta[((size_t)b*TSTP + j)*TSTP + k];
  xres[((size_t)b*DIN + i)*TSTP + k] = acc;
}

__global__ void agc_xres2(const float* __restrict__ X, float* __restrict__ xres)
{
  int tid = blockIdx.x*256 + threadIdx.x;             // (b, r<25, t)
  if (tid >= BSZ*25*TSTP) return;
  int t = tid % TSTP, r = (tid/TSTP) % 25, b = tid/(25*TSTP);
  float v = (r < NNOD) ? X[((size_t)b*NNOD + r)*TSTP*FIN + t*FIN + 0]
                       : X[((size_t)b*NNOD + 0)*TSTP*FIN + t*FIN + 1];
  xres[((size_t)b*DIN + (NNOD*CO) + r)*TSTP + t] = v;
}

__global__ void agc_xin(const float* __restrict__ xres, const float* __restrict__ liw,
                        const float* __restrict__ lib, float* __restrict__ xin)
{
  int tid = blockIdx.x*256 + threadIdx.x;             // (b,d,t)
  if (tid >= BSZ*DIN*TSTP) return;
  int t = tid % TSTP, d = (tid/TSTP) % DIN, b = tid/(TSTP*DIN);
  float acc = lib[t];
  for (int t2 = 0; t2 < TSTP; ++t2)
    acc += xres[((size_t)b*DIN + d)*TSTP + t2]*liw[t*TSTP + t2];
  xin[tid] = acc;
}

// ---------------------------------------------------------------------------
__global__ void __launch_bounds__(256)
agc_beta(const float* __restrict__ hs, const float* __restrict__ ta2w,
         const float* __restrict__ ta2b, float* __restrict__ bp)
{
  __shared__ float red[256];
  int b = blockIdx.x / TSTP, t = blockIdx.x % TSTP;
  float acc = 0.f;
  for (int tt = 0; tt < TSTP; ++tt) {
    const float* hrow = hs + ((size_t)tt*BSZ + b)*HH;
    const float* wrow = ta2w + (size_t)t*(TSTP*HH) + (size_t)tt*HH;
    for (int h = threadIdx.x; h < HH; h += 256) acc += hrow[h]*wrow[h];
  }
  red[threadIdx.x] = acc; __syncthreads();
  for (int st = 128; st > 0; st >>= 1) { if (threadIdx.x < st) red[threadIdx.x] += red[threadIdx.x+st]; __syncthreads(); }
  if (threadIdx.x == 0) bp[b*TSTP + t] = fmaxf(red[0] + ta2b[t], 0.f);
}

__global__ void __launch_bounds__(256)
agc_out(const float* __restrict__ hs, const float* __restrict__ bp,
        const float* __restrict__ xin, const float* __restrict__ ow,
        const float* __restrict__ ob, float* __restrict__ out)
{
  __shared__ float red[256];
  int b = blockIdx.x;
  float beta[TSTP];
  float mx = -1e30f;
  for (int t = 0; t < TSTP; ++t) mx = fmaxf(mx, bp[b*TSTP + t]);
  float sum = 0.f;
  for (int t = 0; t < TSTP; ++t) { beta[t] = expf(bp[b*TSTP + t] - mx); sum += beta[t]; }
  float isum = 1.f/sum;
  for (int t = 0; t < TSTP; ++t) beta[t] *= isum;
  float acc = 0.f;
  for (int h = threadIdx.x; h < HH; h += 256) {
    float v = xin[(size_t)b*HH + h];     // x_in.reshape(B,-1): (B,DIN,T) flat == b*H + h
    for (int t = 0; t < TSTP; ++t) v += hs[((size_t)t*BSZ + b)*HH + h]*beta[t];
    acc += fmaxf(v, 0.f)*ow[h];
  }
  red[threadIdx.x] = acc; __syncthreads();
  for (int st = 128; st > 0; st >>= 1) { if (threadIdx.x < st) red[threadIdx.x] += red[threadIdx.x+st]; __syncthreads(); }
  if (threadIdx.x == 0) out[b] = red[0] + ob[0];
}

static inline unsigned cdiv256(size_t n){ return (unsigned)((n + 255)/256); }

} // namespace

extern "C" void kernel_launch(void* const* d_in, const int* in_sizes, int n_in,
                              void* d_out, int out_size, void* d_ws, size_t ws_size,
                              hipStream_t stream)
{
  (void)in_sizes; (void)n_in; (void)out_size; (void)ws_size;
  const float* X    = (const float*)d_in[0];
  const float* cheb = (const float*)d_in[1];
  const float* saW1 = (const float*)d_in[2];
  const float* saW2 = (const float*)d_in[3];
  const float* saW3 = (const float*)d_in[4];
  const float* sabs = (const float*)d_in[5];
  const float* saVs = (const float*)d_in[6];
  const float* taU1 = (const float*)d_in[7];
  const float* taU2 = (const float*)d_in[8];
  const float* taU3 = (const float*)d_in[9];
  const float* tabe = (const float*)d_in[10];
  const float* taVe = (const float*)d_in[11];
  const float* theta= (const float*)d_in[12];
  const float* chbb = (const float*)d_in[13];
  const float* lng  = (const float*)d_in[14];
  const float* lnb  = (const float*)d_in[15];
  const float* liw  = (const float*)d_in[16];
  const float* lib  = (const float*)d_in[17];
  const float* sa2w = (const float*)d_in[18];
  const float* sa2b = (const float*)d_in[19];
  const float* Wih  = (const float*)d_in[20];
  const float* Whh  = (const float*)d_in[21];
  const float* bih  = (const float*)d_in[22];
  const float* bhh  = (const float*)d_in[23];
  const float* ta2w = (const float*)d_in[24];
  const float* ta2b = (const float*)d_in[25];
  const float* outw = (const float*)d_in[26];
  const float* outb = (const float*)d_in[27];
  const float* h0   = (const float*)d_in[28];
  const float* c0   = (const float*)d_in[29];

  float* out    = (float*)d_out;          // 32 floats
  float* out_sa = out + BSZ;              // 18432 floats (s_a)

  // Workspace carve-up (needs ~135 MB; dominated by bf16-packed weights: 123 MB < 192 MB L2).
  char* p = (char*)d_ws;
  auto take = [&](size_t bytes) -> void* {
    void* r = (void*)p; p += (bytes + 255) & ~(size_t)255; return r;
  };
  unsigned int*   Wpk  = (unsigned int*)  take((size_t)NTI*KT*1024);
  unsigned int*   Apk  = (unsigned int*)  take((size_t)MTI*KT*1024);
  float* Af    = (float*)take(sizeof(float)*(size_t)BSZ*KPAD);
  float* gates = (float*)take(sizeof(float)*(size_t)BSZ*H4);
  float* cbuf  = (float*)take(sizeof(float)*(size_t)BSZ*HH);
  float* hsb   = (float*)take(sizeof(float)*(size_t)TSTP*BSZ*HH);
  float* xin   = (float*)take(sizeof(float)*(size_t)BSZ*DIN*TSTP);
  float* saB   = (float*)take(sizeof(float)*(size_t)BSZ*NNOD*NNOD);
  float* taB   = (float*)take(sizeof(float)*(size_t)BSZ*TSTP*TSTP);
  float* lhsS  = (float*)take(sizeof(float)*(size_t)BSZ*NNOD*TSTP);
  float* rhsS  = (float*)take(sizeof(float)*(size_t)BSZ*TSTP*NNOD);
  float* Qs    = (float*)take(sizeof(float)*(size_t)BSZ*NNOD*NNOD);
  float* Ss    = (float*)take(sizeof(float)*(size_t)BSZ*NNOD*NNOD);
  float* q2s   = (float*)take(sizeof(float)*(size_t)BSZ*NNOD*NNOD);
  float* lhsT  = (float*)take(sizeof(float)*(size_t)BSZ*TSTP*NNOD);
  float* rhsT  = (float*)take(sizeof(float)*(size_t)BSZ*NNOD*TSTP);
  float* QE    = (float*)take(sizeof(float)*(size_t)BSZ*TSTP*TSTP);
  float* Eb    = (float*)take(sizeof(float)*(size_t)BSZ*TSTP*TSTP);
  float* q2t   = (float*)take(sizeof(float)*(size_t)BSZ*TSTP*TSTP);
  float* rgB   = (float*)take(sizeof(float)*(size_t)BSZ*KC*NNOD*FIN*TSTP);
  float* xcB   = (float*)take(sizeof(float)*(size_t)BSZ*NNOD*CO*TSTP);
  float* xgc   = (float*)take(sizeof(float)*(size_t)BSZ*NNOD*CO*TSTP);
  float* xres  = (float*)take(sizeof(float)*(size_t)BSZ*DIN*TSTP);
  float* bpB   = (float*)take(sizeof(float)*(size_t)BSZ*TSTP);

  // Weight pack (once per launch; makes the 12-step weight stream L2-resident).
  agc_packW<<<cdiv256((size_t)NTI*KT*256), 256, 0, stream>>>(Wih, Whh, Wpk);

  // Spatial attention -> s_a (also written to d_out).
  agc_sa1<<<cdiv256(BSZ*NNOD*TSTP), 256, 0, stream>>>(X, saW1, saW2, saW3, lhsS, rhsS);
  agc_sa2<<<cdiv256(BSZ*NNOD*NNOD), 256, 0, stream>>>(lhsS, rhsS, sabs, Qs);
  agc_sa3<<<cdiv256(BSZ*NNOD*NNOD), 256, 0, stream>>>(saVs, Qs, Ss);
  agc_sa4<<<cdiv256(BSZ*NNOD),      256, 0, stream>>>(Ss, q2s);
  agc_sa5<<<cdiv256(NNOD*NNOD),     256, 0, stream>>>(q2s, saB, out_sa);

  // Temporal attention -> t_a.
  agc_ta1<<<cdiv256(BSZ*NNOD*TSTP), 256, 0, stream>>>(X, taU1, taU2, taU3, lhsT, rhsT);
  agc_ta2<<<cdiv256(BSZ*TSTP*TSTP), 256, 0, stream>>>(lhsT, rhsT, tabe, QE);
  agc_ta3<<<cdiv256(BSZ*TSTP*TSTP), 256, 0, stream>>>(taVe, QE, Eb);
  agc_ta4<<<cdiv256(BSZ*TSTP),      256, 0, stream>>>(Eb, q2t);
  agc_ta5<<<cdiv256(TSTP*TSTP),     256, 0, stream>>>(q2t, taB);

  // Graph conv + LN + temporal mix + input projection.
  agc_rg  <<<cdiv256(BSZ*KC*NNOD*FIN*TSTP), 256, 0, stream>>>(X, cheb, saB, rgB);
  agc_gc  <<<cdiv256(BSZ*NNOD*CO*TSTP),     256, 0, stream>>>(rgB, theta, chbb, xcB);
  agc_ln  <<<cdiv256(BSZ*NNOD*CO),          256, 0, stream>>>(xcB, lng, lnb, xgc);
  agc_xgct<<<cdiv256(BSZ*NNOD*CO*TSTP),     256, 0, stream>>>(xgc, taB, xres);
  agc_xres2<<<cdiv256(BSZ*25*TSTP),         256, 0, stream>>>(X, xres);
  agc_xin <<<cdiv256(BSZ*DIN*TSTP),         256, 0, stream>>>(xres, liw, lib, xin);

  // LSTM: 12 sequential steps; heavy WMMA GEMM each step.
  agc_init<<<cdiv256(BSZ*KPAD), 256, 0, stream>>>(h0, c0, Af, cbuf);
  for (int t = 0; t < TSTP; ++t) {
    agc_alpha<<<BSZ, 512, 0, stream>>>(xin, sa2w, sa2b, Af, t);
    agc_packA<<<cdiv256(MTI*KT*256), 256, 0, stream>>>(Af, Apk);
    agc_gates<<<(NTI + 7)/8, 256, 0, stream>>>(
        (const unsigned short*)Apk, (const unsigned short*)Wpk, bih, bhh, gates);
    agc_cell<<<cdiv256(BSZ*HH), 256, 0, stream>>>(gates, cbuf, hsb, Af, t);
  }

  // Output attention + head.
  agc_beta<<<BSZ*TSTP, 256, 0, stream>>>(hsb, ta2w, ta2b, bpB);
  agc_out <<<BSZ,      256, 0, stream>>>(hsb, bpB, xin, outw, outb, out);
}
